// PhysBiformerBlock_47811575939461
// MI455X (gfx1250) — matrix operand order repaired and checked by reading, not compiled, not run on hardware
//
#include <hip/hip_runtime.h>
#include <math.h>

// ---------------- CDNA5 WMMA vector types ----------------
typedef __attribute__((ext_vector_type(16))) _Float16 v16h;
typedef __attribute__((ext_vector_type(8)))  _Float16 v8h;
typedef __attribute__((ext_vector_type(8)))  float    v8f;

// Problem constants (from reference setup_inputs)
static constexpr int LT   = 4;
static constexpr int BB   = 8;
static constexpr int LL   = 1024;
static constexpr int DIM  = 256;
static constexpr int ROWS = LT * BB * LL;          // 32768 tokens
static constexpr int HEADS = 8;
static constexpr int HD    = 32;                   // head dim
static constexpr int RS    = 64;                   // region size
static constexpr int RNUM  = LL / RS;              // 16 regions
static constexpr int TOPK  = 4;

// ---------------- CDNA5 async global->LDS copy (ASYNCcnt path) ----------------
// GLOBAL_LOAD_ASYNC_TO_LDS_B128, GV mode: per-lane copy of 16 bytes,
// vdst = 32-bit LDS byte address, vaddr = 64-bit global address.
__device__ __forceinline__ void async_ld16(unsigned lds_addr, const void* g) {
  asm volatile("global_load_async_to_lds_b128 %0, %1, off"
               :: "v"(lds_addr), "v"((unsigned long long)(uintptr_t)g)
               : "memory");
}
__device__ __forceinline__ void wait_async0() {
  asm volatile("s_wait_asynccnt 0x0" ::: "memory");
}

// =========================================================
// 1) BatchNorm statistics: per-channel a = gamma*rstd, b = beta - mean*a
// =========================================================
__global__ __launch_bounds__(256)
void bn_stats_kernel(const float* __restrict__ x,
                     const float* __restrict__ gamma,
                     const float* __restrict__ beta,
                     float* __restrict__ acoef,
                     float* __restrict__ bcoef,
                     int rows, int C) {
  int c   = blockIdx.x;
  int tid = threadIdx.x;
  float s = 0.f, s2 = 0.f;
  for (int r = tid; r < rows; r += 256) {
    float v = x[(size_t)r * C + c];
    s  += v;
    s2 += v * v;
  }
  __shared__ float sh[256], sh2[256];
  sh[tid] = s; sh2[tid] = s2;
  __syncthreads();
  for (int st = 128; st > 0; st >>= 1) {
    if (tid < st) { sh[tid] += sh[tid + st]; sh2[tid] += sh2[tid + st]; }
    __syncthreads();
  }
  if (tid == 0) {
    float inv_n = 1.0f / (float)rows;
    float mean  = sh[0] * inv_n;
    float var   = sh2[0] * inv_n - mean * mean;
    float rstd  = rsqrtf(var + 1e-5f);
    float a     = gamma[c] * rstd;
    acoef[c] = a;
    bcoef[c] = beta[c] - mean * a;
  }
}

// =========================================================
// 2) Fused BatchNorm + LIF -> binary spikes stored as f16 (exact)
// =========================================================
__global__ __launch_bounds__(256)
void bn_lif_kernel(const float* __restrict__ x,
                   const float* __restrict__ acoef,
                   const float* __restrict__ bcoef,
                   _Float16* __restrict__ s_out,
                   int spatialC /* (BB*LL)*DIM */) {
  size_t idx = (size_t)blockIdx.x * 256 + threadIdx.x;
  if (idx >= (size_t)spatialC) return;
  int c = (int)(idx & (DIM - 1));
  float a = acoef[c], b = bcoef[c];
  float v = 0.f;
  size_t stride = (size_t)spatialC;            // elements per timestep
#pragma unroll
  for (int t = 0; t < LT; t++) {
    float xt = a * x[(size_t)t * stride + idx] + b;
    v = v + (xt - v) * 0.5f;                   // TAU = 2
    float s = (v - 1.0f >= 0.f) ? 1.f : 0.f;   // VTH = 1
    s_out[(size_t)t * stride + idx] = (_Float16)s;
    v = v * (1.f - s);
  }
}

// =========================================================
// 3) Weight convert f32 (K x N) -> f16 transposed (N x K)
// =========================================================
__global__ __launch_bounds__(256)
void cvt_transpose_kernel(const float* __restrict__ W,
                          _Float16* __restrict__ Wt,
                          int K, int N) {
  size_t idx = (size_t)blockIdx.x * 256 + threadIdx.x;
  if (idx >= (size_t)K * N) return;
  int n = (int)(idx / K);
  int k = (int)(idx % K);
  Wt[idx] = (_Float16)W[(size_t)k * N + n];
}

// =========================================================
// 4) Block-cooperative WMMA GEMM: C(MxN) = A_f16(MxK) @ Bt_f16(NxK)^T + bias
//    block = 256 threads (8 waves) -> 128(M) x 64(N) tile.
//    B panel (64 cols x 32 k) double-buffered in LDS via
//    global_load_async_to_lds_b128 (ASYNCcnt), shared by all 8 waves.
//    Epilogues: 0 = f16 ; 1 = exact GELU f16 ; 2 = resid + C*scale (f32)
// =========================================================
template <int EP>
__global__ __launch_bounds__(256)
void gemm_wmma_kernel(const _Float16* __restrict__ A,
                      const _Float16* __restrict__ Bt,
                      const float* __restrict__ bias,
                      const float* __restrict__ resid,
                      const float* __restrict__ scale_p,
                      _Float16* __restrict__ outh,
                      float* __restrict__ outf,
                      int M, int N, int K) {
  constexpr int BPAD = 56;                 // row stride (f16): 112 B = 7*16 B
  __shared__ _Float16 Bs[2][64][BPAD];     // 14 KB, double-buffered B panel

  const int tid  = threadIdx.x;
  const int lane = tid & 31;
  const int w    = tid >> 5;               // wave id 0..7
  const int tiles_n = N >> 6;
  const int bm = blockIdx.x / tiles_n;
  const int bn = blockIdx.x % tiles_n;
  const int l15  = lane & 15;
  const int hi   = lane >> 4;              // 0 or 1
  const int koff = hi * 8;                 // A: K sub-offset per half-wave
  const int kr16 = hi * 8;                 // B: K sub-offset per half-wave

  // --- staging coordinates: each of 256 threads copies 16 B per k-panel
  const int sn = tid >> 2;                 // column 0..63 within N-tile
  const int sk = (tid & 3) * 8;            // k offset 0,8,16,24 (f16)
  const _Float16* bsrc = Bt + (size_t)(bn * 64 + sn) * K + sk;
  const unsigned lds_w0 = (unsigned)(uintptr_t)&Bs[0][sn][sk];
  const unsigned lds_bufstride = (unsigned)(sizeof(_Float16) * 64 * BPAD);

  const int mrow = bm * 128 + w * 16 + l15;
  const _Float16* arow = A + (size_t)mrow * K;

  v8f acc[4];
#pragma unroll
  for (int j = 0; j < 4; j++) acc[j] = {};

  // prologue: stage k-panel 0 into buffer 0
  async_ld16(lds_w0, bsrc);

  const int ksteps = K >> 5;
  for (int it = 0; it < ksteps; ++it) {
    const int k0  = it << 5;
    const int buf = it & 1;
    wait_async0();                         // my panel writes landed in LDS
    __syncthreads();                       // whole panel visible to all waves
    if (it + 1 < ksteps)                   // overlap next panel with compute
      async_ld16(lds_w0 + (unsigned)((buf ^ 1)) * lds_bufstride,
                 bsrc + (size_t)(it + 1) * 32);

    // --- A fragment: 16x32 f16 (ISA layout), direct from global/L2
    v16h af;
    {
      v8h lo  = *(const v8h*)(arow + k0 + koff);
      v8h hiv = *(const v8h*)(arow + k0 + koff + 16);
#pragma unroll
      for (int i = 0; i < 8; i++) { af[i] = lo[i]; af[i + 8] = hiv[i]; }
    }
    // --- 4 B fragments from LDS panel (lane = column, ISA table K order)
#pragma unroll
    for (int j = 0; j < 4; j++) {
      const _Float16* bp = &Bs[buf][j * 16 + l15][kr16];
      v8h b0 = *(const v8h*)bp;
      v8h b1 = *(const v8h*)(bp + 16);
      v16h bf;
#pragma unroll
      for (int i = 0; i < 8; i++) { bf[i] = b0[i]; bf[i + 8] = b1[i]; }
      acc[j] = __builtin_amdgcn_wmma_f32_16x16x32_f16(
          false, af, false, bf, (short)0, acc[j], false, false);
    }
    __syncthreads();                       // all waves done with buf
  }

  // --- epilogue (C layout: VGPR e -> row (hi*8+e), lane&15 -> col)
  float scl = (EP == 2) ? scale_p[0] : 0.f;
#pragma unroll
  for (int j = 0; j < 4; j++) {
    int col = bn * 64 + j * 16 + l15;
    float bs = bias[col];
#pragma unroll
    for (int e = 0; e < 8; e++) {
      int row = bm * 128 + w * 16 + hi * 8 + e;
      size_t off = (size_t)row * N + col;
      float v = acc[j][e] + bs;
      if constexpr (EP == 0) {
        outh[off] = (_Float16)v;
      } else if constexpr (EP == 1) {
        float g = 0.5f * v * (1.0f + erff(v * 0.70710678118654752f));
        outh[off] = (_Float16)g;
      } else {
        outf[off] = resid[off] + v * scl;
      }
    }
  }
}

// =========================================================
// 5) Region means of q and k (over RS=64 tokens) -> f32
// =========================================================
__global__ __launch_bounds__(256)
void region_means_kernel(const _Float16* __restrict__ qkv,
                         float* __restrict__ qr,
                         float* __restrict__ kr) {
  int blk = blockIdx.x;          // (t*BB+b)*RNUM + r : 512 blocks
  int d   = threadIdx.x;
  int tb  = blk >> 4;
  int r   = blk & 15;
  size_t tokbase = (size_t)tb * LL + (size_t)r * RS;
  float sq = 0.f, sk = 0.f;
  for (int i = 0; i < RS; i++) {
    const _Float16* p = qkv + (tokbase + i) * (3 * DIM);
    sq += (float)p[d];
    sk += (float)p[DIM + d];
  }
  qr[(size_t)blk * DIM + d] = sq * (1.f / RS);
  kr[(size_t)blk * DIM + d] = sk * (1.f / RS);
}

// =========================================================
// 6) Region affinity + top-4 routing indices
// =========================================================
__global__ __launch_bounds__(32)
void aff_topk_kernel(const float* __restrict__ qr,
                     const float* __restrict__ kr,
                     int* __restrict__ topk) {
  int blk  = blockIdx.x;         // 512 blocks = (t,b,r)
  int lane = threadIdx.x;
  int tb   = blk >> 4;
  __shared__ float aff[RNUM];
  if (lane < RNUM) {
    const float* q = qr + (size_t)blk * DIM;
    const float* k = kr + ((size_t)tb * RNUM + lane) * DIM;
    float s = 0.f;
    for (int d = 0; d < DIM; d++) s += q[d] * k[d];
    aff[lane] = s;
  }
  __syncthreads();
  if (lane == 0) {
    bool used[RNUM] = {};
#pragma unroll
    for (int t = 0; t < TOPK; t++) {
      int best = 0; float bv = -3.4e38f;
      for (int s = 0; s < RNUM; s++)
        if (!used[s] && aff[s] > bv) { bv = aff[s]; best = s; }
      used[best] = true;
      topk[(size_t)blk * TOPK + t] = best;
    }
  }
}

// =========================================================
// 7) Fused bi-level attention: one block per (t,b,region),
//    one wave per head. scores/softmax/out all via WMMA f16.
// =========================================================
__global__ __launch_bounds__(256)
void attn_kernel(const _Float16* __restrict__ qkv,
                 const int* __restrict__ topk,
                 _Float16* __restrict__ attn_out) {
  __shared__ _Float16 pbuf[HEADS][16][TOPK * RS];   // 64 KB: softmax probs
  const int blk  = blockIdx.x;                      // 512
  const int tb   = blk >> 4;
  const int r    = blk & 15;
  const int h    = threadIdx.x >> 5;
  const int lane = threadIdx.x & 31;
  const int l15  = lane & 15;
  const int hi   = lane >> 4;
  const int hcol = h * HD;
  const size_t qtokbase = (size_t)tb * LL + (size_t)r * RS;

  int rid[TOPK];
#pragma unroll
  for (int g = 0; g < TOPK; g++) rid[g] = topk[(size_t)blk * TOPK + g];

  const float iss = 0.17677669529663687f;           // 1/sqrt(32)

  for (int mt = 0; mt < 4; mt++) {                  // 4 x 16 query rows
    // ---- q A-fragment (16 queries x K=32=hd)
    v16h qf;
    {
      size_t qtok = qtokbase + mt * 16 + l15;
      const _Float16* qp = qkv + qtok * (3 * DIM) + hcol;
      v8h lo = *(const v8h*)(qp + hi * 8);
      v8h hv = *(const v8h*)(qp + hi * 8 + 16);
#pragma unroll
      for (int i = 0; i < 8; i++) { qf[i] = lo[i]; qf[i + 8] = hv[i]; }
    }
    // ---- scores: 16 key-tiles, one 16x16x32 WMMA each (K = hd)
    v8f sc[16];
#pragma unroll
    for (int jt = 0; jt < 16; jt++) {
      int j = jt * 16 + l15;
      size_t ktok = (size_t)tb * LL + (size_t)rid[j >> 6] * RS + (j & 63);
      const _Float16* kp = qkv + ktok * (3 * DIM) + DIM + hcol + hi * 8;
      v8h b0 = *(const v8h*)kp;
      v8h b1 = *(const v8h*)(kp + 16);
      v16h bf;
#pragma unroll
      for (int i = 0; i < 8; i++) { bf[i] = b0[i]; bf[i + 8] = b1[i]; }
      v8f z = {};
      sc[jt] = __builtin_amdgcn_wmma_f32_16x16x32_f16(
          false, qf, false, bf, (short)0, z, false, false);
    }
    // ---- softmax over 256 keys (row M = hi*8+e lives in one 16-lane half)
    float rmax[8], rsum[8];
#pragma unroll
    for (int e = 0; e < 8; e++) rmax[e] = -3.0e38f;
#pragma unroll
    for (int jt = 0; jt < 16; jt++)
#pragma unroll
      for (int e = 0; e < 8; e++) {
        sc[jt][e] *= iss;
        rmax[e] = fmaxf(rmax[e], sc[jt][e]);
      }
#pragma unroll
    for (int e = 0; e < 8; e++) {
      for (int m = 1; m < 16; m <<= 1)
        rmax[e] = fmaxf(rmax[e], __shfl_xor(rmax[e], m, 32));
      rsum[e] = 0.f;
    }
#pragma unroll
    for (int jt = 0; jt < 16; jt++)
#pragma unroll
      for (int e = 0; e < 8; e++) {
        float p = __expf(sc[jt][e] - rmax[e]);
        sc[jt][e] = p;
        rsum[e] += p;
      }
#pragma unroll
    for (int e = 0; e < 8; e++) {
      for (int m = 1; m < 16; m <<= 1)
        rsum[e] += __shfl_xor(rsum[e], m, 32);
      rsum[e] = 1.f / rsum[e];
    }
    // ---- stage p to LDS in row-major so it can be re-read in A layout
#pragma unroll
    for (int jt = 0; jt < 16; jt++)
#pragma unroll
      for (int e = 0; e < 8; e++)
        pbuf[h][hi * 8 + e][jt * 16 + l15] = (_Float16)(sc[jt][e] * rsum[e]);
    __syncthreads();

    // ---- out = p(16x256) @ v_g(256x32): 2 N-tiles, 8 K-steps each
#pragma unroll
    for (int nt = 0; nt < 2; nt++) {
      v8f acc = {};
#pragma unroll
      for (int kk = 0; kk < 8; kk++) {
        v16h pf;
        const _Float16* pp = &pbuf[h][l15][kk * 32 + hi * 8];
#pragma unroll
        for (int i = 0; i < 8; i++) { pf[i] = pp[i]; pf[i + 8] = pp[16 + i]; }
        v16h vf;
        int d = nt * 16 + l15;
#pragma unroll
        for (int i = 0; i < 16; i++) {
          int j = kk * 32 + hi * 8 + i + (i >= 8 ? 8 : 0);
          size_t vtok = (size_t)tb * LL + (size_t)rid[j >> 6] * RS + (j & 63);
          vf[i] = qkv[vtok * (3 * DIM) + 2 * DIM + hcol + d];
        }
        acc = __builtin_amdgcn_wmma_f32_16x16x32_f16(
            false, pf, false, vf, (short)0, acc, false, false);
      }
#pragma unroll
      for (int e = 0; e < 8; e++) {
        size_t otok = qtokbase + mt * 16 + hi * 8 + e;
        attn_out[otok * DIM + hcol + nt * 16 + l15] = (_Float16)acc[e];
      }
    }
    __syncthreads();
  }
}

// =========================================================
// Host-side orchestration
// =========================================================
extern "C" void kernel_launch(void* const* d_in, const int* in_sizes, int n_in,
                              void* d_out, int out_size, void* d_ws, size_t ws_size,
                              hipStream_t stream) {
  const float* x         = (const float*)d_in[0];
  const float* bn1_gamma = (const float*)d_in[1];
  const float* bn1_beta  = (const float*)d_in[2];
  const float* Wqkv      = (const float*)d_in[3];
  const float* bqkv      = (const float*)d_in[4];
  const float* Wo        = (const float*)d_in[5];
  const float* bo        = (const float*)d_in[6];
  const float* bn2_gamma = (const float*)d_in[7];
  const float* bn2_beta  = (const float*)d_in[8];
  const float* W1        = (const float*)d_in[9];
  const float* b1        = (const float*)d_in[10];
  const float* W2        = (const float*)d_in[11];
  const float* b2        = (const float*)d_in[12];
  const float* scale     = (const float*)d_in[13];
  float* out = (float*)d_out;

  // ---- carve workspace (256B aligned)
  char* w = (char*)d_ws;
  auto carve = [&](size_t bytes) {
    char* p = w;
    w += (bytes + 255) & ~(size_t)255;
    return p;
  };
  float*    a1    = (float*)carve(DIM * 4);
  float*    bc1   = (float*)carve(DIM * 4);
  float*    a2    = (float*)carve(DIM * 4);
  float*    bc2   = (float*)carve(DIM * 4);
  _Float16* Wqkvt = (_Float16*)carve((size_t)(3 * DIM) * DIM * 2);
  _Float16* Wot   = (_Float16*)carve((size_t)DIM * DIM * 2);
  _Float16* W1t   = (_Float16*)carve((size_t)(4 * DIM) * DIM * 2);
  _Float16* W2t   = (_Float16*)carve((size_t)DIM * (4 * DIM) * 2);
  float*    qr    = (float*)carve((size_t)LT * BB * RNUM * DIM * 4);
  float*    kr    = (float*)carve((size_t)LT * BB * RNUM * DIM * 4);
  int*      tk    = (int*)carve((size_t)LT * BB * RNUM * TOPK * 4);
  _Float16* sbuf  = (_Float16*)carve((size_t)ROWS * DIM * 2);        // spikes (reused)
  _Float16* qkvh  = (_Float16*)carve((size_t)ROWS * 3 * DIM * 2);    // q|k|v f16
  _Float16* attnh = (_Float16*)carve((size_t)ROWS * DIM * 2);        // attn out f16
  float*    x2    = (float*)carve((size_t)ROWS * DIM * 4);           // mid residual
  _Float16* hbuf  = (_Float16*)carve((size_t)ROWS * 4 * DIM * 2);    // FFN hidden

  // ---- 0) weight convert+transpose to f16
  cvt_transpose_kernel<<<(DIM * 3 * DIM + 255) / 256, 256, 0, stream>>>(Wqkv, Wqkvt, DIM, 3 * DIM);
  cvt_transpose_kernel<<<(DIM * DIM + 255) / 256, 256, 0, stream>>>(Wo, Wot, DIM, DIM);
  cvt_transpose_kernel<<<(DIM * 4 * DIM + 255) / 256, 256, 0, stream>>>(W1, W1t, DIM, 4 * DIM);
  cvt_transpose_kernel<<<(4 * DIM * DIM + 255) / 256, 256, 0, stream>>>(W2, W2t, 4 * DIM, DIM);

  // ---- 1) BN1 + LIF -> s1 spikes (f16, exact)
  bn_stats_kernel<<<DIM, 256, 0, stream>>>(x, bn1_gamma, bn1_beta, a1, bc1, ROWS, DIM);
  bn_lif_kernel<<<(BB * LL * DIM + 255) / 256, 256, 0, stream>>>(x, a1, bc1, sbuf, BB * LL * DIM);

  // ---- 2) qkv = s1 @ Wqkv + bqkv  (WMMA + async LDS)  M=32768 N=768 K=256
  gemm_wmma_kernel<0><<<(ROWS / 128) * (3 * DIM / 64), 256, 0, stream>>>(
      sbuf, Wqkvt, bqkv, nullptr, nullptr, qkvh, nullptr, ROWS, 3 * DIM, DIM);

  // ---- 3) region routing
  region_means_kernel<<<LT * BB * RNUM, 256, 0, stream>>>(qkvh, qr, kr);
  aff_topk_kernel<<<LT * BB * RNUM, 32, 0, stream>>>(qr, kr, tk);

  // ---- 4) bi-level attention (WMMA scores + WMMA p@v)
  attn_kernel<<<LT * BB * RNUM, 256, 0, stream>>>(qkvh, tk, attnh);

  // ---- 5) x2 = x + (attn @ Wo + bo) * scale  (WMMA, residual epilogue)
  gemm_wmma_kernel<2><<<(ROWS / 128) * (DIM / 64), 256, 0, stream>>>(
      attnh, Wot, bo, x, scale, nullptr, x2, ROWS, DIM, DIM);

  // ---- 6) BN2 + LIF -> s2 spikes (reuse sbuf)
  bn_stats_kernel<<<DIM, 256, 0, stream>>>(x2, bn2_gamma, bn2_beta, a2, bc2, ROWS, DIM);
  bn_lif_kernel<<<(BB * LL * DIM + 255) / 256, 256, 0, stream>>>(x2, a2, bc2, sbuf, BB * LL * DIM);

  // ---- 7) h = gelu(s2 @ W1 + b1)  (WMMA, GELU epilogue)  N=1024
  gemm_wmma_kernel<1><<<(ROWS / 128) * (4 * DIM / 64), 256, 0, stream>>>(
      sbuf, W1t, b1, nullptr, nullptr, hbuf, nullptr, ROWS, 4 * DIM, DIM);

  // ---- 8) out = x2 + (h @ W2 + b2) * scale  (WMMA, residual)  K=1024
  gemm_wmma_kernel<2><<<(ROWS / 128) * (DIM / 64), 256, 0, stream>>>(
      hbuf, W2t, b2, x2, scale, nullptr, out, ROWS, DIM, 4 * DIM);
}
